// PairedKidneyBackbone_40939628265811
// MI455X (gfx1250) — matrix operand order repaired and checked by reading, not compiled, not run on hardware
//
#include <hip/hip_runtime.h>
#include <hip/hip_bf16.h>

// ---------------------------------------------------------------------------
// PairedKidneyBackbone on MI455X (gfx1250, wave32):
//   - all 6 dense GEMMs via v_wmma_f32_16x16x32_f16 (f16 in, f32 accum)
//   - 128x64x32 block tile, 8 waves, 32x32 per wave (4 WMMA / 4 frag loads)
//   - double-buffered fragment-major LDS: global loads for tile k+1 issued
//     before the WMMAs of tile k; one barrier per K-step
//   - graph edge phase: wave-per-node score dots, wave-per-edge softmax
//     scatter with global f32 atomics (working set fits in 192MB L2)
// Workspace requirement: ~125 MB.
// ---------------------------------------------------------------------------

typedef __attribute__((ext_vector_type(16))) _Float16 v16h;
typedef __attribute__((ext_vector_type(8)))  _Float16 v8h;
typedef __attribute__((ext_vector_type(4)))  _Float16 v4h;
typedef __attribute__((ext_vector_type(8)))  float    v8f;

#define GEMM_BM 128
#define GEMM_BN 64
#define GEMM_BK 32

#define PK_WMMA(A_, B_, C_) \
    __builtin_amdgcn_wmma_f32_16x16x32_f16(false, (A_), false, (B_), (short)0, (C_), false, false)

// ------------------------------ GEMM (WMMA) --------------------------------
// C[M,N] = A[M,K] @ W[K,N] + bias.
// Fragment-major LDS (this code's contiguous k = 16*g + slot order):
//  A element (r,k): sub=r>>4, lane=(r&15)+16*g, g=k>>4,
//                   slot = k - 16*g
//  B element (k,c): nsub=c>>4, lane=(c&15)+16*g, g=k>>4, slot = k - 16*g
// so each lane reads a whole v16h operand as 32 contiguous bytes.
__global__ __launch_bounds__(256) void pk_gemm_wmma(
    const float* __restrict__ A, const float* __restrict__ W,
    const float* __restrict__ bias, float* __restrict__ C,
    int M, int K, int N)
{
    __shared__ __align__(32) _Float16 As[2][8 * 32 * 16];   // 128x32 per buf
    __shared__ __align__(32) _Float16 Bs[2][4 * 32 * 16];   // 32x64 per buf

    const int tid  = threadIdx.x;
    const int wave = tid >> 5;
    const int lane = tid & 31;
    const int bm = blockIdx.y * GEMM_BM;
    const int bn = blockIdx.x * GEMM_BN;

    // ---- A-tile loader constants: 4 float4 per thread (128x32 f32) ----
    int adst[4], grow[4], acol[4];
#pragma unroll
    for (int i = 0; i < 4; ++i) {
        int idx4 = tid + i * 256;
        int r  = idx4 >> 3;            // 8 float4 per 32-wide row
        int c4 = (idx4 & 7) * 4;
        int g  = c4 >> 4;
        int slot = c4 - 16 * g;
        adst[i] = (r >> 4) * 512 + ((r & 15) + 16 * g) * 16 + slot;
        grow[i] = min(bm + r, M - 1);
        acol[i] = c4;
    }
    // ---- B-tile loader constants: 8 strided b32 -> one b128 store ----
    const int bc  = tid & 63;
    const int bkb = (tid >> 6) * 8;
    const int bg  = bkb >> 4;
    const int bdst = (bc >> 4) * 512 + ((bc & 15) + 16 * bg) * 16 + (bkb - 16 * bg);

    const int wmi = (wave & 3) * 2;    // A subtile base (rows (wave&3)*32)
    const int wni = (wave >> 2) * 2;   // B subtile base (cols (wave>>2)*32)

    v8f acc00 = {}, acc01 = {}, acc10 = {}, acc11 = {};

    float4 fa[4];
    float  fb[8];

    auto gload = [&](int k0) {
#pragma unroll
        for (int i = 0; i < 4; ++i)
            fa[i] = *(const float4*)&A[(long)grow[i] * K + k0 + acol[i]];
        const float* wp = &W[(long)(k0 + bkb) * N + bn + bc];
#pragma unroll
        for (int j = 0; j < 8; ++j) { fb[j] = wp[0]; wp += N; }
    };
    auto sstore = [&](int buf) {
#pragma unroll
        for (int i = 0; i < 4; ++i) {
            v4h h = {(_Float16)fa[i].x, (_Float16)fa[i].y,
                     (_Float16)fa[i].z, (_Float16)fa[i].w};
            *(v4h*)&As[buf][adst[i]] = h;
        }
        v8h hb;
#pragma unroll
        for (int j = 0; j < 8; ++j) hb[j] = (_Float16)fb[j];
        *(v8h*)&Bs[buf][bdst] = hb;
    };

    gload(0);
    sstore(0);
    __syncthreads();

    const int nk = K / GEMM_BK;
    for (int it = 0; it < nk; ++it) {
        const int  cur  = it & 1;
        const bool more = (it + 1) < nk;
        if (more) {
            if (it + 2 < nk) {   // global_prefetch_b8 two tiles ahead
                __builtin_prefetch(&A[(long)grow[0] * K + (it + 2) * GEMM_BK + acol[0]], 0, 1);
                __builtin_prefetch(&W[(long)((it + 2) * GEMM_BK + bkb) * N + bn + bc], 0, 1);
            }
            gload((it + 1) * GEMM_BK);     // loads in flight during WMMAs
        }
        v16h a0 = *(const v16h*)&As[cur][(wmi + 0) * 512 + lane * 16];
        v16h a1 = *(const v16h*)&As[cur][(wmi + 1) * 512 + lane * 16];
        v16h b0 = *(const v16h*)&Bs[cur][(wni + 0) * 512 + lane * 16];
        v16h b1 = *(const v16h*)&Bs[cur][(wni + 1) * 512 + lane * 16];
        acc00 = PK_WMMA(a0, b0, acc00);
        acc01 = PK_WMMA(a0, b1, acc01);
        acc10 = PK_WMMA(a1, b0, acc10);
        acc11 = PK_WMMA(a1, b1, acc11);
        if (more) sstore(cur ^ 1);
        __syncthreads();
    }

    // ---- epilogue: C/D layout lane(0-15)=N, VGPR r -> M = r + 8*(lane>>4) ----
    const int g  = lane >> 4;
    const int lr = lane & 15;
    const int c0 = bn + (wave >> 2) * 32 + lr;
    const int c1 = c0 + 16;
    const float bv0 = bias[c0];
    const float bv1 = bias[c1];
    const int rbase = bm + (wave & 3) * 32 + 8 * g;
    if (bm + GEMM_BM <= M) {           // fast path: full tile, no guards
#pragma unroll
        for (int r = 0; r < 8; ++r) {
            C[(long)(rbase + r) * N + c0]      = acc00[r] + bv0;
            C[(long)(rbase + r) * N + c1]      = acc01[r] + bv1;
            C[(long)(rbase + 16 + r) * N + c0] = acc10[r] + bv0;
            C[(long)(rbase + 16 + r) * N + c1] = acc11[r] + bv1;
        }
    } else {
#pragma unroll
        for (int r = 0; r < 8; ++r) {
            if (rbase + r < M) {
                C[(long)(rbase + r) * N + c0] = acc00[r] + bv0;
                C[(long)(rbase + r) * N + c1] = acc01[r] + bv1;
            }
            if (rbase + 16 + r < M) {
                C[(long)(rbase + 16 + r) * N + c0] = acc10[r] + bv0;
                C[(long)(rbase + 16 + r) * N + c1] = acc11[r] + bv1;
            }
        }
    }
}

// --------------------------- small helper kernels --------------------------

// stage-1 embedding: t1[n,c] = nf[n,:7] @ W1[:,c] + b1[c]   (H=256)
__global__ __launch_bounds__(256) void pk_embed1(
    const float* __restrict__ nf, const float* __restrict__ W1,
    const float* __restrict__ b1, float* __restrict__ t1, int M)
{
    int i = blockIdx.x * 256 + threadIdx.x;
    if (i >= M * 256) return;
    int n = i >> 8, c = i & 255;
    const float* f = nf + (long)n * 7;
    float s = b1[c];
#pragma unroll
    for (int k = 0; k < 7; ++k) s += f[k] * W1[k * 256 + c];
    t1[i] = s;
}

// wave-per-node: ss[n] = h[n,:]·a_s ; sd[n] = h[n,:]·a_d
__global__ __launch_bounds__(256) void pk_node_scores(
    const float* __restrict__ h, const float* __restrict__ a_s,
    const float* __restrict__ a_d, float* __restrict__ ss,
    float* __restrict__ sd, int M, int C)
{
    int w = (blockIdx.x * 256 + threadIdx.x) >> 5;
    int lane = threadIdx.x & 31;
    if (w >= M) return;
    const float* row = h + (long)w * C;
    float s0 = 0.f, s1 = 0.f;
    for (int c = lane; c < C; c += 32) {
        float v = row[c];
        s0 += v * a_s[c];
        s1 += v * a_d[c];
    }
#pragma unroll
    for (int off = 16; off > 0; off >>= 1) {
        s0 += __shfl_down(s0, off, 32);
        s1 += __shfl_down(s1, off, 32);
    }
    if (lane == 0) { ss[w] = s0; sd[w] = s1; }
}

// order-preserving float<->uint for atomicMax segment-max
__device__ __forceinline__ unsigned pk_enc(float f) {
    unsigned u = __float_as_uint(f);
    return (u & 0x80000000u) ? ~u : (u | 0x80000000u);
}
__device__ __forceinline__ float pk_dec(unsigned u) {
    return __uint_as_float((u & 0x80000000u) ? (u & 0x7FFFFFFFu) : ~u);
}

__global__ __launch_bounds__(256) void pk_edge_max(
    const int* __restrict__ src, const int* __restrict__ dst,
    const float* __restrict__ ss, const float* __restrict__ sd,
    float* __restrict__ e, unsigned* __restrict__ m, int E)
{
    int i = blockIdx.x * 256 + threadIdx.x;
    if (i >= E) return;
    float v = ss[src[i]] + sd[dst[i]];
    v = (v > 0.f) ? v : 0.2f * v;            // leaky_relu(0.2)
    e[i] = v;
    atomicMax(&m[dst[i]], pk_enc(v));
}

__global__ __launch_bounds__(256) void pk_edge_exp(
    const int* __restrict__ dst, float* __restrict__ e,
    const unsigned* __restrict__ m, float* __restrict__ denom, int E)
{
    int i = blockIdx.x * 256 + threadIdx.x;
    if (i >= E) return;
    int d = dst[i];
    float ex = __expf(e[i] - pk_dec(m[d]));
    e[i] = ex;
    atomicAdd(&denom[d], ex);
}

// wave-per-edge scatter: out[dst,:] += alpha * h[src,:]
__global__ __launch_bounds__(256) void pk_edge_scatter(
    const int* __restrict__ src, const int* __restrict__ dst,
    const float* __restrict__ e, const float* __restrict__ denom,
    const float* __restrict__ h, float* __restrict__ out, int E, int C)
{
    int w = (blockIdx.x * 256 + threadIdx.x) >> 5;
    int lane = threadIdx.x & 31;
    if (w >= E) return;
    int s = src[w], d = dst[w];
    float alpha = e[w] / denom[d];
    const float* hs = h + (long)s * C;
    float* od = out + (long)d * C;
    for (int c = lane; c < C; c += 32) atomicAdd(&od[c], hs[c] * alpha);
}

// graph LayerNorm pass 1: per-graph sum & sumsq of (g + bias[c])
__global__ __launch_bounds__(256) void pk_ln_reduce(
    const float* __restrict__ gin, const float* __restrict__ b,
    float* __restrict__ gsum, int NC, int C, int total)
{
    __shared__ float s1[256], s2[256];
    int i = blockIdx.x * 256 + threadIdx.x;
    float v = 0.f, v2 = 0.f;
    if (i < total) {
        float x = gin[i] + b[i % C];
        v = x; v2 = x * x;
    }
    s1[threadIdx.x] = v; s2[threadIdx.x] = v2;
    __syncthreads();
#pragma unroll
    for (int st = 128; st > 0; st >>= 1) {
        if (threadIdx.x < st) {
            s1[threadIdx.x] += s1[threadIdx.x + st];
            s2[threadIdx.x] += s2[threadIdx.x + st];
        }
        __syncthreads();
    }
    if (threadIdx.x == 0) {
        int graph = (blockIdx.x * 256) / NC;   // NC divisible by 256
        atomicAdd(&gsum[graph * 2], s1[0]);
        atomicAdd(&gsum[graph * 2 + 1], s2[0]);
    }
}

// graph LayerNorm pass 2 + ReLU
__global__ __launch_bounds__(256) void pk_ln_norm_relu(
    const float* __restrict__ gin, const float* __restrict__ b,
    const float* __restrict__ gsum, const float* __restrict__ lnw,
    const float* __restrict__ lnb, float* __restrict__ out,
    int NC, int C, int total)
{
    int i = blockIdx.x * 256 + threadIdx.x;
    if (i >= total) return;
    int graph = i / NC;
    float cnt  = (float)NC;
    float mu   = gsum[graph * 2] / cnt;
    float var  = gsum[graph * 2 + 1] / cnt - mu * mu;
    float rstd = rsqrtf(var + 1e-5f);
    int c = i % C;
    float x = gin[i] + b[c];
    float y = (x - mu) * rstd * lnw[c] + lnb[c];
    out[i] = (y > 0.f) ? y : 0.f;
}

// x += g + bias[c]   (post-g2 residual into embedding x, C=256)
__global__ __launch_bounds__(256) void pk_residual_bias(
    float* __restrict__ x, const float* __restrict__ g,
    const float* __restrict__ b, int total)
{
    int i = blockIdx.x * 256 + threadIdx.x;
    if (i >= total) return;
    x[i] += g[i] + b[i & 255];
}

// x += relu(t)
__global__ __launch_bounds__(256) void pk_add_relu(
    float* __restrict__ x, const float* __restrict__ t, int total)
{
    int i = blockIdx.x * 256 + threadIdx.x;
    if (i >= total) return;
    float v = t[i];
    x[i] += (v > 0.f) ? v : 0.f;
}

// out = concat( x * act[n] , act )
__global__ __launch_bounds__(256) void pk_finalize(
    const float* __restrict__ x, const float* __restrict__ act,
    float* __restrict__ out, int nt)
{
    int i = blockIdx.x * 256 + threadIdx.x;
    int total = nt * 256;
    if (i < total)              out[i] = x[i] * act[i >> 8];
    else if (i < total + nt)    out[i] = act[i - total];
}

// ------------------------------- host side ---------------------------------

static inline int divup(int a, int b) { return (a + b - 1) / b; }

static void run_gemm(const float* A, const float* W, const float* bias,
                     float* C, int M, int K, int N, hipStream_t stream) {
    dim3 grid(N / GEMM_BN, divup(M, GEMM_BM));
    pk_gemm_wmma<<<grid, 256, 0, stream>>>(A, W, bias, C, M, K, N);
}

extern "C" void kernel_launch(void* const* d_in, const int* in_sizes, int n_in,
                              void* d_out, int out_size, void* d_ws, size_t ws_size,
                              hipStream_t stream) {
    (void)n_in; (void)out_size; (void)ws_size;
    const float* nf    = (const float*)d_in[0];
    const float* act   = (const float*)d_in[1];
    const int*   src   = (const int*)d_in[2];
    const int*   dst   = (const int*)d_in[3];
    const float* embW1 = (const float*)d_in[4];
    const float* embb1 = (const float*)d_in[5];
    const float* embW2 = (const float*)d_in[6];
    const float* embb2 = (const float*)d_in[7];
    const float* gW[3]  = {(const float*)d_in[8],  (const float*)d_in[14], (const float*)d_in[20]};
    const float* gAS[3] = {(const float*)d_in[9],  (const float*)d_in[15], (const float*)d_in[21]};
    const float* gAD[3] = {(const float*)d_in[10], (const float*)d_in[16], (const float*)d_in[22]};
    const float* gB[3]  = {(const float*)d_in[11], (const float*)d_in[17], (const float*)d_in[23]};
    const float* lnW[2] = {(const float*)d_in[12], (const float*)d_in[18]};
    const float* lnB[2] = {(const float*)d_in[13], (const float*)d_in[19]};
    const float* ff1W = (const float*)d_in[24];
    const float* ff1b = (const float*)d_in[25];
    const float* ff2W = (const float*)d_in[26];
    const float* ff2b = (const float*)d_in[27];

    const int nt = in_sizes[1];          // B*N = 20000
    const int Et = in_sizes[2];          // total edges incl self loops = 340000
    const int Bn = 2;                    // graphs (reference constant)
    const int H = 256, H2 = 512;
    const int NCgraph = (nt / Bn);       // nodes per graph

    // ---- workspace carve-up (floats) ----
    float* x   = (float*)d_ws;                    // nt*H   embedding / running x
    float* t   = x + (size_t)nt * H;              // nt*H   temp (emb stage1 / FF)
    float* bufA = t + (size_t)nt * H;             // nt*H2
    float* bufB = bufA + (size_t)nt * H2;         // nt*H2
    float* ss  = bufB + (size_t)nt * H2;          // nt
    float* sd  = ss + nt;                         // nt
    unsigned* m = (unsigned*)(sd + nt);           // nt
    float* denom = (float*)m + nt;                // nt
    float* e   = denom + nt;                      // Et
    float* gsum = e + Et;                         // 2*Bn
    float* zbias = gsum + 2 * Bn;                 // H2 zeros (bias stub)

    hipMemsetAsync(zbias, 0, (size_t)H2 * sizeof(float), stream);

    // ---- embedding: x = (nf @ W1 + b1) @ W2 + b2 ----
    pk_embed1<<<divup(nt * H, 256), 256, 0, stream>>>(nf, embW1, embb1, t, nt);
    run_gemm(t, embW2, embb2, x, nt, H, H, stream);

    // ---- GAT layers ----
    // layer dims: 0: 256->512 (LN0), 1: 512->512 (LN1), 2: 512->256 (no LN)
    const float* layer_in[3] = {x, bufA, bufB};
    float* hbuf[3]  = {bufA, bufB, bufA};         // GEMM output (= h)
    float* aggbuf[3]= {bufB, bufA, bufB};         // scatter accumulation
    const int Cin[3]  = {H, H2, H2};
    const int Cout[3] = {H2, H2, H};

    for (int L = 0; L < 3; ++L) {
        const float* xin = layer_in[L];
        float* h  = hbuf[L];
        float* gg = aggbuf[L];
        int Ci = Cin[L], Co = Cout[L];

        run_gemm(xin, gW[L], zbias, h, nt, Ci, Co, stream);
        pk_node_scores<<<divup(nt, 8), 256, 0, stream>>>(h, gAS[L], gAD[L], ss, sd, nt, Co);

        hipMemsetAsync(m, 0, (size_t)nt * sizeof(unsigned), stream);
        hipMemsetAsync(denom, 0, (size_t)nt * sizeof(float), stream);
        hipMemsetAsync(gg, 0, (size_t)nt * Co * sizeof(float), stream);

        pk_edge_max<<<divup(Et, 256), 256, 0, stream>>>(src, dst, ss, sd, e, m, Et);
        pk_edge_exp<<<divup(Et, 256), 256, 0, stream>>>(dst, e, m, denom, Et);
        pk_edge_scatter<<<divup(Et, 8), 256, 0, stream>>>(src, dst, e, denom, h, gg, Et, Co);

        if (L < 2) {
            int total = nt * Co;
            hipMemsetAsync(gsum, 0, 2 * Bn * sizeof(float), stream);
            pk_ln_reduce<<<divup(total, 256), 256, 0, stream>>>(
                gg, gB[L], gsum, NCgraph * Co, Co, total);
            pk_ln_norm_relu<<<divup(total, 256), 256, 0, stream>>>(
                gg, gB[L], gsum, lnW[L], lnB[L], h, NCgraph * Co, Co, total);
        } else {
            pk_residual_bias<<<divup(nt * H, 256), 256, 0, stream>>>(x, gg, gB[L], nt * H);
        }
    }

    // ---- residual FF layers ----
    run_gemm(x, ff1W, ff1b, t, nt, H, H, stream);
    pk_add_relu<<<divup(nt * H, 256), 256, 0, stream>>>(x, t, nt * H);
    run_gemm(x, ff2W, ff2b, t, nt, H, H, stream);
    pk_add_relu<<<divup(nt * H, 256), 256, 0, stream>>>(x, t, nt * H);

    // ---- mask + pack outputs (x*mask, active_agents) ----
    pk_finalize<<<divup(nt * H + nt, 256), 256, 0, stream>>>(x, act, (float*)d_out, nt);
}
